// SelfAttentionFinal_43130061586978
// MI455X (gfx1250) — matrix operand order repaired and checked by reading, not compiled, not run on hardware
//
#include <hip/hip_runtime.h>
#include <math.h>

// ---------- types ----------
typedef __attribute__((ext_vector_type(16))) __bf16    v16bf;
typedef __attribute__((ext_vector_type(8)))  __bf16    v8bf;
typedef __attribute__((ext_vector_type(8)))  float     v8f;
typedef __attribute__((ext_vector_type(4)))  unsigned  u32x4;
typedef __attribute__((ext_vector_type(8)))  int       i32x8;
typedef __attribute__((ext_vector_type(4)))  int       i32x4;

__device__ __forceinline__ unsigned short f2bfu(float f) {
    union { float f; unsigned u; } x; x.f = f;
    unsigned u = x.u + 0x7FFFu + ((x.u >> 16) & 1u);   // round-to-nearest-even
    return (unsigned short)(u >> 16);
}

// ---------- fp32 -> bf16 conversion ----------
__global__ __launch_bounds__(256)
void cvt_f32_bf16_kernel(const float* __restrict__ x, unsigned short* __restrict__ y, long n) {
    long i = (long)blockIdx.x * blockDim.x + threadIdx.x;
    long stride = (long)gridDim.x * blockDim.x;
    for (; i < n; i += stride) y[i] = f2bfu(x[i]);
}

// ---------- fp32 -> bf16 with transpose: y[c][r] = x[r][c], x is RxC ----------
__global__ __launch_bounds__(256)
void cvt_f32_bf16_t_kernel(const float* __restrict__ x, unsigned short* __restrict__ y,
                           int R, int C) {
    long n = (long)R * C;
    long i = (long)blockIdx.x * blockDim.x + threadIdx.x;
    long stride = (long)gridDim.x * blockDim.x;
    for (; i < n; i += stride) {
        long r = i / C, c = i - r * C;
        y[c * (long)R + r] = f2bfu(x[i]);
    }
}

// ---------- WMMA GEMM: C(MxN) = scale * A(MxK) * B(NxK row-major)^T + bias ----------
// Block tile 128x128, K-step 32, 256 threads (8 waves), wave tile 32x64.
// A tile staged by the Tensor Data Mover (tensor_load_to_lds, TENSORcnt);
// B tile staged by per-lane global_load_async_to_lds_b128 (ASYNCcnt).
// LDS rows padded to 40 shorts (80B): TDM pad feature inserts 4 dwords per 16 dwords.
template<bool OUT_BF16, bool BIAS, bool STORE_T>
__global__ __launch_bounds__(256)
void gemm_wmma_kernel(const unsigned short* __restrict__ A, long lda, long strideA,
                      const unsigned short* __restrict__ B, long ldb, long strideB,
                      void* __restrict__ Cv, long ldc, long strideC,
                      const float* __restrict__ bias, int Kdim, float scale) {
    __shared__ unsigned short sA[128][40];
    __shared__ unsigned short sB[128][40];

    const int t    = threadIdx.x;
    const int lane = t & 31;
    const int wave = t >> 5;
    const int waveM = wave & 3;            // 4 waves along M (32 rows each)
    const int waveN = wave >> 2;           // 2 waves along N (64 cols each)
    const int g  = lane >> 4;              // K-half selector
    const int hl = lane & 15;

    const long m0 = (long)blockIdx.y * 128;
    const long n0 = (long)blockIdx.x * 128;
    const long z  = blockIdx.z;
    A += z * strideA;
    B += z * strideB;

    const unsigned ldsA = (unsigned)(unsigned long long)&sA[0][0];

    const v8f vzero = {0.f,0.f,0.f,0.f,0.f,0.f,0.f,0.f};
    v8f c[2][4];
    #pragma unroll
    for (int mi = 0; mi < 2; ++mi)
        #pragma unroll
        for (int nj = 0; nj < 4; ++nj) c[mi][nj] = vzero;

    // B staging addresses are loop-invariant except for k0
    const int srow = t >> 1;               // 0..127  (n index)
    const int scol = (t & 1) << 4;         // 0 or 16 (k index)
    const unsigned ldsB = (unsigned)(unsigned long long)&sB[srow][scol];

    for (int k0 = 0; k0 < Kdim; k0 += 32) {
        // ---- A tile: one TDM descriptor, issued by wave 0 ----
        if (wave == 0) {
            unsigned long long ga =
                (unsigned long long)(const void*)(A + m0 * lda + k0);
            unsigned ldau = (unsigned)lda;
            u32x4 g0 = { 1u,                       // count=1, user descriptor
                         ldsA,                     // lds_addr
                         (unsigned)ga,             // global_addr[31:0]
                         (unsigned)((ga >> 32) & 0x01FFFFFFu) | 0x80000000u }; // addr[56:32] | type=2
            i32x8 g1 = { (int)((1u << 16)          // data_size = 2 bytes
                              | (1u << 20)         // pad_enable
                              | (3u << 22)         // pad_interval: 16 dwords (64B)
                              | (3u << 25)),       // pad_amount:   4 dwords (16B)
                         (int)((ldau & 0xFFFFu) << 16),            // tensor_dim0[15:0]
                         (int)((ldau >> 16) & 0xFFFFu),            // tensor_dim0[31:16], dim1 lo=0
                         (int)(0x10u | (32u << 16)),               // tensor_dim1 hi (1<<20), tile_dim0=32
                         (int)128,                                 // tile_dim1=128, tile_dim2=0
                         (int)ldau,                                // tensor_dim0_stride[31:0]
                         0, 0 };                                   // stride hi / dim1_stride
            i32x4 zv4 = { 0, 0, 0, 0 };
            i32x8 zv8 = { 0, 0, 0, 0, 0, 0, 0, 0 };
            __builtin_amdgcn_tensor_load_to_lds(g0, g1, zv4, zv4, zv8, 0);
            __builtin_amdgcn_s_wait_tensorcnt(0);
        }

        // ---- B tile: per-lane async copies (16B x2, shared IOFFSET on both sides) ----
        {
            unsigned long long gb =
                (unsigned long long)(const void*)(B + (n0 + srow) * ldb + k0 + scol);
            asm volatile(
                "global_load_async_to_lds_b128 %0, %1, off\n\t"
                "global_load_async_to_lds_b128 %0, %1, off offset:16"
                :: "v"(ldsB), "v"(gb) : "memory");
            asm volatile("s_wait_asynccnt 0" ::: "memory");
        }
        __syncthreads();

        // ---- fragments ----
        v16bf a[2], b[4];
        #pragma unroll
        for (int mi = 0; mi < 2; ++mi) {
            int row = waveM * 32 + mi * 16 + hl;
            v8bf lo = *reinterpret_cast<const v8bf*>(&sA[row][g * 8]);
            v8bf hi = *reinterpret_cast<const v8bf*>(&sA[row][16 + g * 8]);
            a[mi] = __builtin_shufflevector(lo, hi, 0,1,2,3,4,5,6,7,8,9,10,11,12,13,14,15);
        }
        #pragma unroll
        for (int nj = 0; nj < 4; ++nj) {
            int col = waveN * 64 + nj * 16 + hl;
            v8bf lo = *reinterpret_cast<const v8bf*>(&sB[col][g * 8]);
            v8bf hi = *reinterpret_cast<const v8bf*>(&sB[col][16 + g * 8]);
            b[nj] = __builtin_shufflevector(lo, hi, 0,1,2,3,4,5,6,7,8,9,10,11,12,13,14,15);
        }

        // ---- 8 WMMAs per wave per K-step ----
        #pragma unroll
        for (int mi = 0; mi < 2; ++mi)
            #pragma unroll
            for (int nj = 0; nj < 4; ++nj)
                c[mi][nj] = __builtin_amdgcn_wmma_f32_16x16x32_bf16(
                    false, a[mi], false, b[nj], (short)0, c[mi][nj], false, false);

        __syncthreads();
    }

    // ---- epilogue: C VGPR layout: m = r + 8*g, n = hl ----
    #pragma unroll
    for (int mi = 0; mi < 2; ++mi) {
        #pragma unroll
        for (int nj = 0; nj < 4; ++nj) {
            long colg = n0 + waveN * 64 + nj * 16 + hl;
            float bv = 0.f;
            if constexpr (BIAS) bv = bias[colg];
            #pragma unroll
            for (int r = 0; r < 8; ++r) {
                long rowg = m0 + waveM * 32 + mi * 16 + g * 8 + r;
                long idx  = STORE_T ? (z * strideC + colg * ldc + rowg)
                                    : (z * strideC + rowg * ldc + colg);
                float v = c[mi][nj][r] * scale + bv;
                if constexpr (OUT_BF16)
                    reinterpret_cast<unsigned short*>(Cv)[idx] = f2bfu(v);
                else
                    reinterpret_cast<float*>(Cv)[idx] = v;
            }
        }
    }
}

// ---------- row softmax: fp32 S -> bf16 P ----------
__global__ __launch_bounds__(256)
void softmax_rows_kernel(const float* __restrict__ S, unsigned short* __restrict__ P, int L) {
    const long row = blockIdx.x;
    const float* s = S + row * (long)L;
    unsigned short* p = P + row * (long)L;
    const int n = L / 256;                  // 8 elements per thread (L = 2048)
    float e[8];

    float m = -3.402823466e38f;
    for (int i = 0; i < n; ++i) m = fmaxf(m, s[(long)i * 256 + threadIdx.x]);
    #pragma unroll
    for (int o = 16; o >= 1; o >>= 1) m = fmaxf(m, __shfl_xor(m, o, 32));
    __shared__ float redm[8];
    __shared__ float reds[8];
    if ((threadIdx.x & 31) == 0) redm[threadIdx.x >> 5] = m;
    __syncthreads();
    m = redm[0];
    #pragma unroll
    for (int i = 1; i < 8; ++i) m = fmaxf(m, redm[i]);

    float sum = 0.f;
    for (int i = 0; i < n; ++i) {
        float v = __expf(s[(long)i * 256 + threadIdx.x] - m);
        e[i] = v;
        sum += v;
    }
    #pragma unroll
    for (int o = 16; o >= 1; o >>= 1) sum += __shfl_xor(sum, o, 32);
    if ((threadIdx.x & 31) == 0) reds[threadIdx.x >> 5] = sum;
    __syncthreads();
    sum = 0.f;
    #pragma unroll
    for (int i = 0; i < 8; ++i) sum += reds[i];
    float rinv = 1.f / sum;
    for (int i = 0; i < n; ++i) p[(long)i * 256 + threadIdx.x] = f2bfu(e[i] * rinv);
}

// ---------- launch ----------
extern "C" void kernel_launch(void* const* d_in, const int* in_sizes, int n_in,
                              void* d_out, int out_size, void* d_ws, size_t ws_size,
                              hipStream_t stream) {
    (void)in_sizes; (void)n_in; (void)out_size; (void)ws_size;
    constexpr int Bn = 4, Sq = 2048, D = 1024;
    constexpr long BS = (long)Bn * Sq;          // 8192 rows
    const float* X  = (const float*)d_in[0];
    const float* Wq = (const float*)d_in[1];
    const float* bq = (const float*)d_in[2];
    const float* Wk = (const float*)d_in[3];
    const float* bk = (const float*)d_in[4];
    const float* Wv = (const float*)d_in[5];
    const float* bv = (const float*)d_in[6];
    float* Out = (float*)d_out;

    // workspace layout (bytes)
    char* ws = (char*)d_ws;
    const size_t MiB = 1u << 20;
    unsigned short* Qb  = (unsigned short*)(ws + 0 * MiB);    // 16 MiB (8192 x 1024)
    unsigned short* Kb  = (unsigned short*)(ws + 16 * MiB);   // 16 MiB (8192 x 1024)
    unsigned short* VTb = (unsigned short*)(ws + 32 * MiB);   // 16 MiB (1024 x 8192, V^T)
    unsigned short* Xb  = (unsigned short*)(ws + 48 * MiB);   // 16 MiB
    unsigned short* WqT = (unsigned short*)(ws + 64 * MiB);   // 2 MiB (W^T: f x d)
    unsigned short* WkT = (unsigned short*)(ws + 66 * MiB);   // 2 MiB
    unsigned short* WvT = (unsigned short*)(ws + 68 * MiB);   // 2 MiB
    float*          Sm  = (float*)         (ws + 70 * MiB);   // 64 MiB (4x2048x2048 f32)
    unsigned short* Pm  = (unsigned short*)(ws + 134 * MiB);  // 32 MiB (4x2048x2048 bf16)

    // 0) operand conversion (weights transposed so every GEMM B is NxK row-major)
    cvt_f32_bf16_kernel  <<<4096, 256, 0, stream>>>(X,  Xb,  BS * D);
    cvt_f32_bf16_t_kernel<<<1024, 256, 0, stream>>>(Wq, WqT, D, D);
    cvt_f32_bf16_t_kernel<<<1024, 256, 0, stream>>>(Wk, WkT, D, D);
    cvt_f32_bf16_t_kernel<<<1024, 256, 0, stream>>>(Wv, WvT, D, D);

    // 1) QKV projections: (8192x1024)*(1024x1024)+bias; V stored transposed (d x seq)
    {
        dim3 grid(D / 128, BS / 128, 1);
        gemm_wmma_kernel<true, true, false><<<grid, 256, 0, stream>>>(
            Xb, D, 0, WqT, D, 0, Qb, D, 0, bq, D, 1.0f);
        gemm_wmma_kernel<true, true, false><<<grid, 256, 0, stream>>>(
            Xb, D, 0, WkT, D, 0, Kb, D, 0, bk, D, 1.0f);
        gemm_wmma_kernel<true, true, true><<<grid, 256, 0, stream>>>(
            Xb, D, 0, WvT, D, 0, VTb, BS, 0, bv, D, 1.0f);
    }

    // 2) S = Q * K^T / sqrt(D) per batch
    {
        dim3 grid(Sq / 128, Sq / 128, Bn);
        gemm_wmma_kernel<false, false, false><<<grid, 256, 0, stream>>>(
            Qb, D, (long)Sq * D,
            Kb, D, (long)Sq * D,
            Sm, Sq, (long)Sq * Sq,
            nullptr, D, 0.03125f);          // 1/sqrt(1024)
    }

    // 3) row softmax -> bf16 P
    softmax_rows_kernel<<<Bn * Sq, 256, 0, stream>>>(Sm, Pm, Sq);

    // 4) Out = P * V per batch; B = V^T (d x seq) is NxK with ldb=8192, batch stride 2048
    {
        dim3 grid(D / 128, Sq / 128, Bn);
        gemm_wmma_kernel<false, false, false><<<grid, 256, 0, stream>>>(
            Pm, Sq, (long)Sq * Sq,
            VTb, BS, (long)Sq,
            Out, D, (long)Sq * D,
            nullptr, Sq, 1.0f);
    }
}
